// LocallyGroupedAttnRPE_33028298506539
// MI455X (gfx1250) — matrix operand order repaired and checked by reading, not compiled, not run on hardware
//
#include <hip/hip_runtime.h>

typedef __bf16 bf16;
typedef __attribute__((ext_vector_type(16))) __bf16 v16bf;
typedef __attribute__((ext_vector_type(8)))  __bf16 v8bf;
typedef __attribute__((ext_vector_type(8)))  float  v8f;
typedef __attribute__((ext_vector_type(4)))  float  v4f;

#define C_DIM 256
#define NH    8
#define HD    32
#define TOK   64           // tokens per 8x8 window
#define NWIN  2048         // 8 batches * 256 windows
#define SCALE 0.17677669529663687f  // 32^-0.5, folded into q

static __device__ __forceinline__ v8f wmma_bf16(v16bf a, v16bf b, v8f c) {
  return __builtin_amdgcn_wmma_f32_16x16x32_bf16(false, a, false, b, (short)0, c,
                                                 false, false);
}

static __device__ __forceinline__ v16bf cat8(v8bf lo, v8bf hi) {
  v16bf r;
#pragma unroll
  for (int i = 0; i < 8; ++i) { r[i] = lo[i]; r[i + 8] = hi[i]; }
  return r;
}

static __device__ __forceinline__ v8f zero8() {
  v8f z;
#pragma unroll
  for (int i = 0; i < 8; ++i) z[i] = 0.f;
  return z;
}

// swizzled index into a [64][256] bf16 LDS tile; 8-element (16B) chunk XOR
// swizzle so the 16 lanes of an A-fragment load hit distinct bank quads.
static __device__ __forceinline__ int swz(int p, int c) {
  int chunk = c >> 3;
  chunk = (chunk & ~15) | ((chunk ^ p) & 15);
  return (p << 8) | (chunk << 3) | (c & 7);
}

// ---------------------------------------------------------------------------
// Kernel 0: convert the four 256x256 weights fp32->bf16 and build the
// sinusoidal PE table (64 tokens x 256 channels, identical for every window).
// ---------------------------------------------------------------------------
__global__ __launch_bounds__(256) void prep_kernel(
    const float* __restrict__ Wq, const float* __restrict__ Wk,
    const float* __restrict__ Wv, const float* __restrict__ Wp,
    bf16* __restrict__ wq, bf16* __restrict__ wk,
    bf16* __restrict__ wv, bf16* __restrict__ wp, bf16* __restrict__ pe) {
  int idx = blockIdx.x * 256 + threadIdx.x;
  if (idx < 4 * 65536) {
    int sel = idx >> 16, off = idx & 65535;
    const float* src = (sel == 0) ? Wq : (sel == 1) ? Wk : (sel == 2) ? Wv : Wp;
    bf16* dst        = (sel == 0) ? wq : (sel == 1) ? wk : (sel == 2) ? wv : wp;
    dst[off] = (bf16)src[off];
  } else {
    int j = idx - 4 * 65536;
    if (j < TOK * C_DIM) {
      int p = j >> 8, c = j & 255;
      int tx = p & 7, ty = p >> 3;
      int grp = c >> 6, f = c & 63;
      float coord = (grp < 2) ? (float)tx : (float)ty;
      float ang = 3.14f * coord * (float)f * (1.0f / 200.0f);
      pe[j] = (bf16)((grp & 1) ? cosf(ang) : sinf(ang));
    }
  }
}

// ---------------------------------------------------------------------------
// Kernel 1: fused q/k/v projection. One block per window (64 tokens).
// LDS holds bf16 tiles of x (for v) and x+pe (for q,k); 8 waves, each owns a
// 32-wide slice of output channels and runs 4 M-tiles x 2 N-tiles x 8 K-steps
// of v_wmma_f32_16x16x32_bf16 per weight matrix.
// ---------------------------------------------------------------------------
__global__ __launch_bounds__(256) void qkv_kernel(
    const float* __restrict__ x,
    const bf16* __restrict__ wq, const bf16* __restrict__ wk,
    const bf16* __restrict__ wv,
    const float* __restrict__ bq, const float* __restrict__ bk,
    const float* __restrict__ bv,
    const bf16* __restrict__ pe,
    bf16* __restrict__ qw, bf16* __restrict__ kw, bf16* __restrict__ vw) {
  __shared__ bf16 xq[TOK * C_DIM];   // x + pe
  __shared__ bf16 xv[TOK * C_DIM];   // x
  int tid = threadIdx.x;
  int lane = tid & 31, wave = tid >> 5;
  int wb = blockIdx.x;
  int b = wb >> 8, win = wb & 255;
  int wy = win >> 4, wx = win & 15;

  // ---- stage window into LDS ----
  {
    int p = tid >> 2;
    int c0 = (tid & 3) * 64;
    int y = wy * 8 + (p >> 3), xc = wx * 8 + (p & 7);
    const float* xrow = x + (((size_t)b * 16384 + y * 128 + xc) * C_DIM);
#pragma unroll
    for (int c = c0; c < c0 + 64; c += 8) {
      v4f f0 = *(const v4f*)(xrow + c);
      v4f f1 = *(const v4f*)(xrow + c + 4);
      v8bf pv = *(const v8bf*)(pe + p * C_DIM + c);
      v8bf nv, qv;
#pragma unroll
      for (int i = 0; i < 4; ++i) {
        nv[i]     = (bf16)f0[i];
        nv[i + 4] = (bf16)f1[i];
        qv[i]     = (bf16)(f0[i] + (float)pv[i]);
        qv[i + 4] = (bf16)(f1[i] + (float)pv[i + 4]);
      }
      *(v8bf*)(xv + swz(p, c)) = nv;
      *(v8bf*)(xq + swz(p, c)) = qv;
    }
  }
  __syncthreads();

  int n0 = wave * 32;
  int lm = lane & 15;
  int aoff = (lane < 16) ? 0 : 16;
  int boff = (lane < 16) ? 0 : 16;
  int rowadd = (lane < 16) ? 0 : 8;

  for (int sel = 0; sel < 3; ++sel) {
    const bf16* lds  = (sel == 2) ? xv : xq;
    const bf16* W    = (sel == 0) ? wq : (sel == 1) ? wk : wv;
    const float* bia = (sel == 0) ? bq : (sel == 1) ? bk : bv;
    float bias0 = bia[n0 + lm];
    float bias1 = bia[n0 + 16 + lm];
    v8f acc[4][2];
#pragma unroll
    for (int mt = 0; mt < 4; ++mt) { acc[mt][0] = zero8(); acc[mt][1] = zero8(); }

    for (int k0 = 0; k0 < C_DIM; k0 += 32) {
      v16bf a[4];
#pragma unroll
      for (int mt = 0; mt < 4; ++mt) {
        int p = mt * 16 + lm;
        a[mt] = cat8(*(const v8bf*)(lds + swz(p, k0 + aoff)),
                     *(const v8bf*)(lds + swz(p, k0 + aoff + 8)));
      }
      v16bf b0 = *(const v16bf*)(W + (size_t)(n0 + lm) * C_DIM + k0 + boff);
      v16bf b1 = *(const v16bf*)(W + (size_t)(n0 + 16 + lm) * C_DIM + k0 + boff);
#pragma unroll
      for (int mt = 0; mt < 4; ++mt) {
        acc[mt][0] = wmma_bf16(a[mt], b0, acc[mt][0]);
        acc[mt][1] = wmma_bf16(a[mt], b1, acc[mt][1]);
      }
    }
#pragma unroll
    for (int mt = 0; mt < 4; ++mt) {
#pragma unroll
      for (int nt = 0; nt < 2; ++nt) {
        int n = n0 + nt * 16 + lm;
        int head = n >> 5, d = n & 31;
        size_t base = ((size_t)wb * NH + head) * (TOK * HD);
        float bb = nt ? bias1 : bias0;
#pragma unroll
        for (int r = 0; r < 8; ++r) {
          int p = mt * 16 + r + rowadd;
          float val = acc[mt][nt][r] + bb;
          if (sel == 0)      qw[base + p * HD + d]  = (bf16)(val * SCALE);
          else if (sel == 1) kw[base + p * HD + d]  = (bf16)val;
          else               vw[base + d * TOK + p] = (bf16)val;  // d-major!
        }
      }
    }
  }
}

// ---------------------------------------------------------------------------
// Kernel 2: attention. One block per (window, head): 4 waves, each owns 16
// query rows. scores = q@k^T (4 WMMAs/wave), LDS softmax, o = attn@v
// (4 WMMAs/wave). q already carries the 1/sqrt(hd) scale.
// ---------------------------------------------------------------------------
__global__ __launch_bounds__(128) void attn_kernel(
    const bf16* __restrict__ qw, const bf16* __restrict__ kw,
    const bf16* __restrict__ vw, bf16* __restrict__ ow) {
  __shared__ float sc[TOK * 68];
  __shared__ bf16  at[TOK * 72];
  int tid = threadIdx.x;
  int lane = tid & 31, wave = tid >> 5;
  size_t base = (size_t)blockIdx.x * (TOK * HD);
  const bf16* q = qw + base;
  const bf16* k = kw + base;
  const bf16* v = vw + base;

  int lm = lane & 15;
  int aoff = (lane < 16) ? 0 : 16;
  int boff = (lane < 16) ? 0 : 16;
  int rowadd = (lane < 16) ? 0 : 8;
  int m0 = wave * 16;

  // ---- scores ----
  {
    int p = m0 + lm;
    v16bf aq = cat8(*(const v8bf*)(q + p * HD + aoff),
                    *(const v8bf*)(q + p * HD + aoff + 8));
#pragma unroll
    for (int nt = 0; nt < 4; ++nt) {
      v16bf bb = *(const v16bf*)(k + (nt * 16 + lm) * HD + boff);
      v8f s = wmma_bf16(aq, bb, zero8());
#pragma unroll
      for (int r = 0; r < 8; ++r)
        sc[(m0 + r + rowadd) * 68 + nt * 16 + lm] = s[r];
    }
  }
  __syncthreads();

  // ---- row softmax (64 rows handled by threads 0..63) ----
  if (tid < TOK) {
    float* row = sc + tid * 68;
    float mx = -1e30f;
#pragma unroll
    for (int j = 0; j < TOK; ++j) mx = fmaxf(mx, row[j]);
    float sum = 0.f;
#pragma unroll
    for (int j = 0; j < TOK; ++j) { float e = __expf(row[j] - mx); row[j] = e; sum += e; }
    float rinv = 1.f / sum;
    bf16* arow = at + tid * 72;
#pragma unroll
    for (int j = 0; j < TOK; ++j) arow[j] = (bf16)(row[j] * rinv);
  }
  __syncthreads();

  // ---- o = attn @ v ----
  v8f o[2] = { zero8(), zero8() };
#pragma unroll
  for (int k0 = 0; k0 < TOK; k0 += 32) {
    int p = m0 + lm;
    v16bf aa = cat8(*(const v8bf*)(at + p * 72 + k0 + aoff),
                    *(const v8bf*)(at + p * 72 + k0 + aoff + 8));
#pragma unroll
    for (int nt = 0; nt < 2; ++nt) {
      v16bf bb = *(const v16bf*)(v + (nt * 16 + lm) * TOK + k0 + boff);
      o[nt] = wmma_bf16(aa, bb, o[nt]);
    }
  }
  bf16* outp = ow + base;
#pragma unroll
  for (int nt = 0; nt < 2; ++nt)
#pragma unroll
    for (int r = 0; r < 8; ++r)
      outp[(m0 + r + rowadd) * HD + nt * 16 + lm] = (bf16)o[nt][r];
}

// ---------------------------------------------------------------------------
// Kernel 3: output projection. One block per window; K-step 32 == one head's
// contiguous [64][32] slab of o. Writes fp32 result back in [B, N, C] order.
// ---------------------------------------------------------------------------
__global__ __launch_bounds__(256) void proj_kernel(
    const bf16* __restrict__ ow, const bf16* __restrict__ wp,
    const float* __restrict__ bp, float* __restrict__ out) {
  int tid = threadIdx.x;
  int lane = tid & 31, wave = tid >> 5;
  int wb = blockIdx.x;
  int b = wb >> 8, win = wb & 255;
  int wy = win >> 4, wx = win & 15;
  size_t obase = (size_t)wb * NH * (TOK * HD);
  int n0 = wave * 32;
  int lm = lane & 15;
  int aoff = (lane < 16) ? 0 : 16;
  int boff = (lane < 16) ? 0 : 16;
  int rowadd = (lane < 16) ? 0 : 8;

  v8f acc[4][2];
#pragma unroll
  for (int mt = 0; mt < 4; ++mt) { acc[mt][0] = zero8(); acc[mt][1] = zero8(); }

  for (int k0 = 0; k0 < C_DIM; k0 += 32) {
    const bf16* ohead = ow + obase + (size_t)(k0 >> 5) * (TOK * HD);
    v16bf a[4];
#pragma unroll
    for (int mt = 0; mt < 4; ++mt) {
      int p = mt * 16 + lm;
      a[mt] = cat8(*(const v8bf*)(ohead + p * HD + aoff),
                   *(const v8bf*)(ohead + p * HD + aoff + 8));
    }
    v16bf b0 = *(const v16bf*)(wp + (size_t)(n0 + lm) * C_DIM + k0 + boff);
    v16bf b1 = *(const v16bf*)(wp + (size_t)(n0 + 16 + lm) * C_DIM + k0 + boff);
#pragma unroll
    for (int mt = 0; mt < 4; ++mt) {
      acc[mt][0] = wmma_bf16(a[mt], b0, acc[mt][0]);
      acc[mt][1] = wmma_bf16(a[mt], b1, acc[mt][1]);
    }
  }
  float bias0 = bp[n0 + lm], bias1 = bp[n0 + 16 + lm];
#pragma unroll
  for (int mt = 0; mt < 4; ++mt)
#pragma unroll
    for (int nt = 0; nt < 2; ++nt) {
      int n = n0 + nt * 16 + lm;
      float bb = nt ? bias1 : bias0;
#pragma unroll
      for (int r = 0; r < 8; ++r) {
        int p = mt * 16 + r + rowadd;
        int y = wy * 8 + (p >> 3), xc = wx * 8 + (p & 7);
        out[(((size_t)b * 16384) + y * 128 + xc) * C_DIM + n] = acc[mt][nt][r] + bb;
      }
    }
}

// ---------------------------------------------------------------------------
extern "C" void kernel_launch(void* const* d_in, const int* in_sizes, int n_in,
                              void* d_out, int out_size, void* d_ws, size_t ws_size,
                              hipStream_t stream) {
  (void)in_sizes; (void)n_in; (void)out_size; (void)ws_size;
  const float* x  = (const float*)d_in[0];
  const float* Wq = (const float*)d_in[1];
  const float* bq = (const float*)d_in[2];
  const float* Wk = (const float*)d_in[3];
  const float* bk = (const float*)d_in[4];
  const float* Wv = (const float*)d_in[5];
  const float* bv = (const float*)d_in[6];
  const float* Wp = (const float*)d_in[7];
  const float* bp = (const float*)d_in[8];
  float* out = (float*)d_out;

  char* ws = (char*)d_ws;
  bf16* wq = (bf16*)(ws);                 // 4 x 128 KB bf16 weights
  bf16* wk = (bf16*)(ws + 131072);
  bf16* wv = (bf16*)(ws + 262144);
  bf16* wp = (bf16*)(ws + 393216);
  bf16* pe = (bf16*)(ws + 524288);        // 32 KB PE table
  const size_t qkv_elems = (size_t)NWIN * NH * TOK * HD;  // 33.5M bf16 each
  bf16* qw = (bf16*)(ws + 557056);
  bf16* kw = qw + qkv_elems;
  bf16* vw = kw + qkv_elems;
  bf16* ow = vw + qkv_elems;

  prep_kernel<<<1088, 256, 0, stream>>>(Wq, Wk, Wv, Wp, wq, wk, wv, wp, pe);
  qkv_kernel<<<NWIN, 256, 0, stream>>>(x, wq, wk, wv, bq, bk, bv, pe, qw, kw, vw);
  attn_kernel<<<NWIN * NH, 128, 0, stream>>>(qw, kw, vw, ow);
  proj_kernel<<<NWIN, 256, 0, stream>>>(ow, wp, bp, out);
}
